// ARSSD_25847113187740
// MI455X (gfx1250) — matrix operand order repaired and checked by reading, not compiled, not run on hardware
//
#include <hip/hip_runtime.h>
#include <hip/hip_bf16.h>
#include <stdint.h>

typedef __attribute__((ext_vector_type(16))) _Float16 v16h;
typedef __attribute__((ext_vector_type(8)))  float    v8f;

#define TSTEPS 16
#define NBATCH 8
#define CIO    12
#define HID    64
#define OG     256          // 4*HIDDEN gate channels
#define BN_EPS 1e-5f
#define ANCH_TOT 2688
#define TNN (TSTEPS*NBATCH)
#define LOC_ELEMS ((size_t)TNN*ANCH_TOT*4)

// ---------------------------------------------------------------------------
// Zero-fill (graph-capture-safe init of h/c state in workspace)
// ---------------------------------------------------------------------------
__global__ void ARSSD_zero_kernel(uint32_t* __restrict__ p, size_t n) {
  for (size_t i = (size_t)blockIdx.x * blockDim.x + threadIdx.x; i < n;
       i += (size_t)gridDim.x * blockDim.x)
    p[i] = 0u;
}

// ---------------------------------------------------------------------------
// Weight pre-pack: w[O=256][cin][3][3] f32  ->  Bpack[(Kpad/16)][256][16] f16
// WINDOW-MAJOR K ordering: k = r*cin + ci (r = kh*3+kw). This makes the
// im2col A-staging math nearly free (r is uniform per 32-K chunk when
// cin==64). Fragment-native: a B-fragment lane (N=o, half) reads its 16
// consecutive K values as one contiguous 32-byte chunk.
// ---------------------------------------------------------------------------
__global__ void ARSSD_pack_kernel(const float* __restrict__ w,
                                  _Float16* __restrict__ bp,
                                  int cin, int nb16) {
  int total = nb16 * OG * 16;
  int kr = cin * 9;
  for (int i = blockIdx.x * blockDim.x + threadIdx.x; i < total;
       i += gridDim.x * blockDim.x) {
    int kb = i / (OG * 16);
    int rr = i - kb * (OG * 16);
    int o  = rr >> 4;
    int kk = rr & 15;
    int k  = kb * 16 + kk;
    float v = 0.f;
    if (k < kr) {
      int r  = k / cin;          // kernel-window position 0..8
      int ci = k - r * cin;      // input channel
      v = w[(o * cin + ci) * 9 + r];
    }
    bp[i] = (_Float16)v;
  }
}

// ---------------------------------------------------------------------------
// Fused ConvLSTM cell step (+ optional fused 1x1 box head with decode).
//  - templated on spatial size HH (div/mod become shifts) and cell kind
//  - block = 128 threads (4 waves, wave32), tile = 16 pixels x 256 channels
//  - x-path GEMM -> fold BN affine + ReLU into accumulators -> h-path GEMM
//    accumulates on top (WMMA D = A*B + C), so only 32 acc VGPRs live
//  - window-major K: for cin=64 paths r=kb>>1 is uniform per chunk, staging
//    address = per-thread base + scalar delta (few VALU per element)
//  - register-resident LSTM pointwise (gate tiles per wave chosen so the
//    i/f/g/o of one cell channel land in the same lane & register slot)
// ---------------------------------------------------------------------------
template <int HH, bool CELL2>
__global__ __launch_bounds__(128)
void ARSSD_cell_kernel(const float* __restrict__ xf32,      // cell1 x-path src (f32, NCHW)
                       const _Float16* __restrict__ xf16,   // cell2 x-path src (f16, [C][P])
                       const _Float16* __restrict__ bpX,    // packed Wx
                       const _Float16* __restrict__ bpH,    // packed Wh
                       const _Float16* __restrict__ hin,    // h_{t-1}  f16 [HID][P]
                       _Float16* __restrict__ hout,         // h_t      f16 [HID][P]
                       float* __restrict__ cst,             // c state  f32 [HID][P] (in place)
                       const float* __restrict__ gmm, const float* __restrict__ bet,
                       const float* __restrict__ mea, const float* __restrict__ var,
                       const float* __restrict__ boxw,      // [12][64] (CELL2 only)
                       float* __restrict__ outp,            // d_out (loc | cls)
                       int t, int anchOff) {
  constexpr int WW   = HH;
  constexpr int HWc  = HH * WW;            // power of two
  constexpr int P    = NBATCH * HWc;
  constexpr int NBX  = CELL2 ? 18 : 4;     // x-path K/32 chunks (Kpad: 576 / 128)
  constexpr int NBH  = 18;                 // h-path: 576/32 exactly

  __shared__ _Float16 As[2][512];          // fragment-native A tiles (16 px x 32 K)
  __shared__ _Float16 Hb[16 * HID];        // h tile for fused box head

  const int tid  = threadIdx.x;
  const int lane = tid & 31;
  const int wv   = tid >> 5;               // wave id 0..3
  const int half = lane >> 4;
  const int lm   = lane & 15;
  const int ptile = blockIdx.x * 16;
  const int colBase = wv * 16 + lm;        // this lane's output channel (mod 64)

  // ---- per-thread staging coordinates (loop-invariant): 4 elems/thread ----
  int skk[4], ssl[4], sy[4], sx[4], sbase[4], sxb[4];
#pragma unroll
  for (int i = 0; i < 4; ++i) {
    int e  = tid + 128 * i;
    int m  = e >> 5, kk = e & 31;
    int hs = kk >> 4;                             // fragment half this K lands in
    int q  = kk & 15;                             // slot within the 16-value run
    skk[i] = kk;
    ssl[i] = m * 32 + hs * 16 + q;                // LDS slot (fragment-native)
    int p  = ptile + m;
    int n  = p / HWc, rem = p - n * HWc;          // shifts (HWc power of two)
    int y  = rem / WW, x = rem - (rem / WW) * WW;
    sy[i] = y; sx[i] = x;
    sbase[i] = kk * P + (n * HH + y) * WW + x;    // [C][P] f16 sources (ci part added later)
    sxb[i]   = n * (CIO * HWc) + y * WW + x;      // NCHW f32 x (cell1)
  }

  // ---- stage from a [64][P] f16 source, window-major K, cin=64 ----
  auto stage64 = [&](const _Float16* __restrict__ src, int bsel, int kb) {
    int r  = kb >> 1;                      // uniform per chunk (SALU)
    int dy = r / 3 - 1, dx = r - (r / 3) * 3 - 1;
    int doff = (kb & 1) * (32 * P) + dy * WW + dx;   // scalar address delta
#pragma unroll
    for (int i = 0; i < 4; ++i) {
      int yy = sy[i] + dy, xx = sx[i] + dx;
      float v = 0.f;
      if (yy >= 0 && yy < HH && xx >= 0 && xx < WW)
        v = (float)src[sbase[i] + doff];
      As[bsel][ssl[i]] = (_Float16)v;
    }
  };

  // ---- stage cell1 x-path from NCHW f32, cin=12 (only 4 chunks) ----
  auto stage12 = [&](int bsel, int kb) {
#pragma unroll
    for (int i = 0; i < 4; ++i) {
      int k = kb * 32 + skk[i];
      float v = 0.f;
      if (k < 108) {
        int r  = k / 12;                   // const-div -> mul/shift
        int ci = k - r * 12;
        int dy = r / 3 - 1, dx = r - (r / 3) * 3 - 1;
        int yy = sy[i] + dy, xx = sx[i] + dx;
        if (yy >= 0 && yy < HH && xx >= 0 && xx < WW)
          v = xf32[sxb[i] + ci * HWc + dy * WW + dx];
      }
      As[bsel][ssl[i]] = (_Float16)v;
    }
  };

  auto stageX = [&](int bsel, int kb) {
    if constexpr (CELL2) stage64(xf16, bsel, kb);
    else                 stage12(bsel, kb);
  };

  v8f acc[4] = {};
  const int fragOff = lm * 32 + half * 16;       // contiguous 32B A fragment

  auto mm = [&](int bsel, int kb, const _Float16* __restrict__ bp) {
    v16h a = *(const v16h*)&As[bsel][fragOff];
    const _Float16* bpk = bp + (size_t)(kb * 2 + half) * OG * 16;
#pragma unroll
    for (int g = 0; g < 4; ++g) {
      v16h b = *(const v16h*)&bpk[(g * 64 + colBase) * 16];
      acc[g] = __builtin_amdgcn_wmma_f32_16x16x32_f16(
          false, a, false, b, (short)0, acc[g], false, false);
    }
  };

  // -------- x-path GEMM (double-buffered; last iter preloads h chunk 0) ----
  stageX(0, 0);
  __syncthreads();
  int buf = 0;
#pragma unroll 1
  for (int kb = 0; kb < NBX; ++kb) {
    if (kb + 1 < NBX) stageX(buf ^ 1, kb + 1);
    else              stage64(hin, buf ^ 1, 0);
    mm(buf, kb, bpX);
    __syncthreads();
    buf ^= 1;
  }

  // -------- fold BN affine + ReLU into accumulators (C of h-path WMMAs) ----
#pragma unroll
  for (int g = 0; g < 4; ++g) {
    int o = g * 64 + colBase;
    float sc = gmm[o] * rsqrtf(var[o] + BN_EPS);
    float sh = bet[o] - mea[o] * sc;
#pragma unroll
    for (int r = 0; r < 8; ++r) {
      float gx = acc[g][r] * sc + sh;
      acc[g][r] = gx > 0.f ? gx : 0.f;
    }
  }

  // -------- h-path GEMM accumulating on top --------
#pragma unroll 1
  for (int kb = 0; kb < NBH; ++kb) {
    if (kb + 1 < NBH) stage64(hin, buf ^ 1, kb + 1);
    mm(buf, kb, bpH);
    __syncthreads();
    buf ^= 1;
  }

  // -------- register-resident LSTM pointwise update --------
  const int ch = colBase;                        // cell channel owned by lane
#pragma unroll
  for (int r = 0; r < 8; ++r) {
    int p = ptile + r + 8 * half;                // pixel M = r + 8*half
    size_t idx = (size_t)ch * P + p;
    float co = cst[idx];
    float ig = 1.f / (1.f + __expf(-acc[0][r]));
    float fg = 1.f / (1.f + __expf(-acc[1][r]));
    float gg = tanhf(acc[2][r]);
    float og = 1.f / (1.f + __expf(-acc[3][r]));
    float cn = fg * co + ig * gg;
    float hn = og * tanhf(cn);
    cst[idx] = cn;
    _Float16 h16 = (_Float16)hn;
    hout[idx] = h16;
    if constexpr (CELL2) Hb[(r + 8 * half) * HID + ch] = h16;
  }

  // -------- fused 1x1 box head + decode (cell2 only) --------
  if constexpr (CELL2) {
    __syncthreads();
#pragma unroll 1
    for (int e = tid; e < 16 * CIO; e += 128) {
      int m = e / CIO, o = e - m * CIO;
      float accb = 0.f;
#pragma unroll
      for (int c2 = 0; c2 < HID; ++c2)
        accb += (float)Hb[m * HID + c2] * boxw[o * HID + c2];
      int p = ptile + m;
      int n = p / HWc, rem = p - n * HWc;
      int tn = t * NBATCH + n;
      int aa = o / 6, j = o - aa * 6;
      int ga = anchOff + aa * HWc + rem;
      if (j < 4) outp[((size_t)tn * ANCH_TOT + ga) * 4 + j] = accb;
      else       outp[LOC_ELEMS + ((size_t)tn * ANCH_TOT + ga) * 2 + (j - 4)] = accb;
    }
  }
}

// ---------------------------------------------------------------------------
// Host launcher
// ---------------------------------------------------------------------------
extern "C" void kernel_launch(void* const* d_in, const int* in_sizes, int n_in,
                              void* d_out, int out_size, void* d_ws, size_t ws_size,
                              hipStream_t stream) {
  (void)out_size; (void)ws_size;
  // ---- identify inputs by size pattern (robust to pytree flattening order)
  const int XS[3] = {TSTEPS * NBATCH * CIO * 32 * 32,
                     TSTEPS * NBATCH * CIO * 16 * 16,
                     TSTEPS * NBATCH * CIO * 8 * 8};
  const float* xin[3] = {nullptr, nullptr, nullptr};
  int plist[64]; int np = 0;
  for (int i = 0; i < n_in; ++i) {
    if      (in_sizes[i] == XS[0] && !xin[0]) xin[0] = (const float*)d_in[i];
    else if (in_sizes[i] == XS[1] && !xin[1]) xin[1] = (const float*)d_in[i];
    else if (in_sizes[i] == XS[2] && !xin[2]) xin[2] = (const float*)d_in[i];
    else if (np < 64) plist[np++] = i;
  }
  struct TP { const float *wx1,*wh1,*g1,*b1,*m1,*v1,*wx2,*wh2,*g2,*b2,*m2,*v2,*bw; };
  TP tp[3];
  for (int s = 0; s < 3; ++s) {
    const int* gi = &plist[s * 13];
    auto PP = [&](int j) { return (const float*)d_in[gi[j]]; };
    if (in_sizes[gi[0]] == 27648) {
      // insertion order: cell1{wx,wh,gamma,beta,mean,var}, cell2{...}, box_w
      tp[s] = {PP(0), PP(1), PP(2), PP(3), PP(4), PP(5),
               PP(6), PP(7), PP(8), PP(9), PP(10), PP(11), PP(12)};
    } else {
      // jax-pytree sorted: box_w, cell1{beta,gamma,mean,var,wh,wx}, cell2{...}
      tp[s] = {PP(6), PP(5), PP(2), PP(1), PP(3), PP(4),
               PP(12), PP(11), PP(8), PP(7), PP(9), PP(10), PP(0)};
    }
  }

  // ---- workspace layout
  uint8_t* wp = (uint8_t*)d_ws;
  size_t off = 0;
  auto alloc = [&](size_t bytes) -> void* {
    off = (off + 255) & ~(size_t)255;
    void* r = wp + off; off += bytes; return r;
  };
  const int Hs[3] = {32, 16, 8};
  struct TW { _Float16 *bx1,*bh1,*bx2,*bh2,*h1a,*h1b,*h2a,*h2b; float *c1,*c2; int H,P; };
  TW tw[3];
  for (int s = 0; s < 3; ++s) {
    int H = Hs[s], P = NBATCH * H * H;
    tw[s].H = H; tw[s].P = P;
    tw[s].bx1 = (_Float16*)alloc((size_t)8  * OG * 16 * 2);   // Kpad=128
    tw[s].bh1 = (_Float16*)alloc((size_t)36 * OG * 16 * 2);   // K=576
    tw[s].bx2 = (_Float16*)alloc((size_t)36 * OG * 16 * 2);
    tw[s].bh2 = (_Float16*)alloc((size_t)36 * OG * 16 * 2);
    tw[s].h1a = (_Float16*)alloc((size_t)HID * P * 2);
    tw[s].h1b = (_Float16*)alloc((size_t)HID * P * 2);
    tw[s].h2a = (_Float16*)alloc((size_t)HID * P * 2);
    tw[s].h2b = (_Float16*)alloc((size_t)HID * P * 2);
    tw[s].c1  = (float*)alloc((size_t)HID * P * 4);
    tw[s].c2  = (float*)alloc((size_t)HID * P * 4);
  }

  // ---- zero initial states (every call; deterministic)
  for (int s = 0; s < 3; ++s) {
    size_t nh = (size_t)HID * tw[s].P / 2;   // f16 elems -> u32 words
    size_t nc = (size_t)HID * tw[s].P;       // f32 elems -> u32 words
    int gh = (int)((nh + 255) / 256), gc = (int)((nc + 255) / 256);
    ARSSD_zero_kernel<<<gh, 256, 0, stream>>>((uint32_t*)tw[s].h1a, nh);
    ARSSD_zero_kernel<<<gh, 256, 0, stream>>>((uint32_t*)tw[s].h2a, nh);
    ARSSD_zero_kernel<<<gc, 256, 0, stream>>>((uint32_t*)tw[s].c1, nc);
    ARSSD_zero_kernel<<<gc, 256, 0, stream>>>((uint32_t*)tw[s].c2, nc);
  }

  // ---- pack weights to f16 fragment-native, window-major-K layout
  for (int s = 0; s < 3; ++s) {
    int t1 = 8 * OG * 16, t2 = 36 * OG * 16;
    ARSSD_pack_kernel<<<(t1 + 255) / 256, 256, 0, stream>>>(tp[s].wx1, tw[s].bx1, CIO, 8);
    ARSSD_pack_kernel<<<(t2 + 255) / 256, 256, 0, stream>>>(tp[s].wh1, tw[s].bh1, HID, 36);
    ARSSD_pack_kernel<<<(t2 + 255) / 256, 256, 0, stream>>>(tp[s].wx2, tw[s].bx2, HID, 36);
    ARSSD_pack_kernel<<<(t2 + 255) / 256, 256, 0, stream>>>(tp[s].wh2, tw[s].bh2, HID, 36);
  }

  // ---- sequential scan over T; 2 fused cell kernels per tracker per step
  const int AOFF[3] = {0, 2048, 2560};
  float* outF = (float*)d_out;
  for (int t = 0; t < TSTEPS; ++t) {
    for (int s = 0; s < 3; ++s) {
      TW& w = tw[s]; TP& p = tp[s];
      _Float16* h1i = (t & 1) ? w.h1b : w.h1a;
      _Float16* h1o = (t & 1) ? w.h1a : w.h1b;
      _Float16* h2i = (t & 1) ? w.h2b : w.h2a;
      _Float16* h2o = (t & 1) ? w.h2a : w.h2b;
      const float* xt = xin[s] + (size_t)t * NBATCH * CIO * w.H * w.H;
      dim3 grid(w.P / 16), blk(128);
      switch (w.H) {
        case 32:
          ARSSD_cell_kernel<32, false><<<grid, blk, 0, stream>>>(
              xt, nullptr, w.bx1, w.bh1, h1i, h1o, w.c1,
              p.g1, p.b1, p.m1, p.v1, nullptr, nullptr, t, 0);
          ARSSD_cell_kernel<32, true><<<grid, blk, 0, stream>>>(
              nullptr, h1o, w.bx2, w.bh2, h2i, h2o, w.c2,
              p.g2, p.b2, p.m2, p.v2, p.bw, outF, t, AOFF[s]);
          break;
        case 16:
          ARSSD_cell_kernel<16, false><<<grid, blk, 0, stream>>>(
              xt, nullptr, w.bx1, w.bh1, h1i, h1o, w.c1,
              p.g1, p.b1, p.m1, p.v1, nullptr, nullptr, t, 0);
          ARSSD_cell_kernel<16, true><<<grid, blk, 0, stream>>>(
              nullptr, h1o, w.bx2, w.bh2, h2i, h2o, w.c2,
              p.g2, p.b2, p.m2, p.v2, p.bw, outF, t, AOFF[s]);
          break;
        default:
          ARSSD_cell_kernel<8, false><<<grid, blk, 0, stream>>>(
              xt, nullptr, w.bx1, w.bh1, h1i, h1o, w.c1,
              p.g1, p.b1, p.m1, p.v1, nullptr, nullptr, t, 0);
          ARSSD_cell_kernel<8, true><<<grid, blk, 0, stream>>>(
              nullptr, h1o, w.bx2, w.bh2, h2i, h2o, w.c2,
              p.g2, p.b2, p.m2, p.v2, p.bw, outF, t, AOFF[s]);
          break;
      }
    }
  }
}
